// GCN_69097433858700
// MI455X (gfx1250) — matrix operand order repaired and checked by reading, not compiled, not run on hardware
//
#include <hip/hip_runtime.h>
#include <hip/hip_bf16.h>
#include <stdint.h>
#include <stddef.h>

// ---------------------------------------------------------------------------
// GCN on MI455X (gfx1250): bf16 WMMA (16x16x32) for the dense x@W GEMMs with
// all N-tiles accumulated per wave (A-fragment reuse), compile-time (K,N)
// so B loads use one base pointer + immediate offsets (no spills), and a
// fused self-loop epilogue; 4-edge-per-wave prefetched atomic scatter for
// the L2-resident (A+I) D^-1/2 aggregation (the dominant phase).
// ---------------------------------------------------------------------------

typedef __attribute__((ext_vector_type(16))) __bf16 v16bf;
typedef __attribute__((ext_vector_type(8)))  __bf16 v8bf;
typedef __attribute__((ext_vector_type(8)))  float  v8f;

__device__ __forceinline__ unsigned short f32_to_bf16_bits(float f) {
  unsigned int u = __builtin_bit_cast(unsigned int, f);
  unsigned int r = 0x7FFFu + ((u >> 16) & 1u);          // round-to-nearest-even
  return (unsigned short)((u + r) >> 16);
}

// ------------------------------ utility kernels ----------------------------

__global__ void k_fill(float* __restrict__ p, float v, int n) {
  int i = blockIdx.x * blockDim.x + threadIdx.x;
  if (i < n) p[i] = v;
}

__global__ void k_degree(const int* __restrict__ ei_row, int E,
                         float* __restrict__ deg) {
  int e = blockIdx.x * blockDim.x + threadIdx.x;
  if (e < E) atomicAdd(&deg[ei_row[e]], 1.0f);
}

__global__ void k_dinv(const float* __restrict__ deg, float* __restrict__ dinv, int n) {
  int i = blockIdx.x * blockDim.x + threadIdx.x;
  if (i < n) dinv[i] = rsqrtf(deg[i]);                  // deg >= 1 (self loop)
}

__global__ void k_cvt_bf16(const float* __restrict__ src,
                           unsigned short* __restrict__ dst, int n) {
  int i = blockIdx.x * blockDim.x + threadIdx.x;
  if (i < n) dst[i] = f32_to_bf16_bits(src[i]);
}

// W[K,N] f32 (row major) -> Wt[N,K] bf16 (row major) so B fragments load
// as contiguous 32B per lane.
__global__ void k_transpose_cvt(const float* __restrict__ W,
                                unsigned short* __restrict__ Wt, int K, int N) {
  int idx = blockIdx.x * blockDim.x + threadIdx.x;      // over N*K
  if (idx >= N * K) return;
  int n = idx / K;
  int k = idx - n * K;
  Wt[idx] = f32_to_bf16_bits(W[(size_t)k * N + n]);
}

// ------------------------------ WMMA GEMM ----------------------------------
// One wave computes a full 16 x N stripe (NT = N/16 accumulators), reusing a
// single A fragment across NT WMMAs per K-step.  K and NT are compile-time,
// so every B-fragment load is base + immediate offset (single address pair).
// Epilogue writes h_pre = A@W and h_agg = dinv^2 * h_pre (self-loop term).
// M = 100000 = 16*6250 exactly, so no row guards.
// Fragment layouts used here (wave32, contiguous k, same for A and B):
//   A 16x32 bf16 : lane (half,r): elems 0..7 -> K = half*16+e,
//                                 elems 8..15 -> K = half*16+e, row M=r
//   B 32x16 bf16 : lane (half,r): elem e -> K = half*16+e, col N=r
//   C 16x16 f32  : elem v -> row M = half*8+v, col N=r
template <int NT, int KK>
__global__ void k_wmma_gemm_fused(const unsigned short* __restrict__ A,
                                  const unsigned short* __restrict__ Bt,
                                  const float* __restrict__ dinv,
                                  float* __restrict__ h_pre,
                                  float* __restrict__ h_agg,
                                  int Mtiles) {
  const int N    = NT * 16;
  const int wave = threadIdx.x >> 5;                    // 0..3
  const int tm   = blockIdx.x * 4 + wave;               // 16-row tile index
  if (tm >= Mtiles) return;
  const int lane = threadIdx.x & 31;
  const int half = lane >> 4;
  const int r    = lane & 15;

  const unsigned short* arow  = A  + (size_t)(tm * 16 + r) * KK;
  const unsigned short* bbase = Bt + (size_t)r * KK + half * 16;   // one pointer

  v8f acc[NT];
#pragma unroll
  for (int t = 0; t < NT; ++t) acc[t] = (v8f){};

#pragma unroll
  for (int k0 = 0; k0 < KK; k0 += 32) {
    v8bf a_lo = *(const v8bf*)(arow + k0 + half * 16);       // K = k0+half*16 .. +7
    v8bf a_hi = *(const v8bf*)(arow + k0 + half * 16 + 8);   // K = k0+half*16+8 ..
    v16bf a;
#pragma unroll
    for (int i = 0; i < 8; ++i) { a[i] = a_lo[i]; a[8 + i] = a_hi[i]; }
#pragma unroll
    for (int t = 0; t < NT; ++t) {
      // t*16*KK and k0 are compile-time constants -> immediate offsets
      v16bf b = *(const v16bf*)(bbase + t * 16 * KK + k0);
      acc[t] = __builtin_amdgcn_wmma_f32_16x16x32_bf16(
          /*neg_a=*/false, a, /*neg_b=*/false, b,
          /*c_mod=*/(short)0, acc[t], /*reuse_a=*/false, /*reuse_b=*/false);
    }
  }

  // self-loop scale per output row
  float dv2[8];
#pragma unroll
  for (int v = 0; v < 8; ++v) {
    float d = dinv[tm * 16 + half * 8 + v];
    dv2[v] = d * d;
  }
  float* prow = h_pre + (size_t)(tm * 16 + half * 8) * N + r;
  float* grow = h_agg + (size_t)(tm * 16 + half * 8) * N + r;
#pragma unroll
  for (int t = 0; t < NT; ++t) {
#pragma unroll
    for (int v = 0; v < 8; ++v) {
      float val = acc[t][v];
      prow[v * N + t * 16] = val;                       // immediate offsets
      grow[v * N + t * 16] = val * dv2[v];
    }
  }
}

// ------------------------------ aggregation --------------------------------

// 4 edges per wave with prefetch: agg[row,:] += dinv[row]*dinv[col]*h[col,:]
__global__ void k_edge_scatter(const int* __restrict__ ei, int E,
                               const float* __restrict__ h,
                               const float* __restrict__ dinv,
                               float* __restrict__ agg, int dim) {
  const int EPW  = 4;
  int wave = blockIdx.x * (blockDim.x >> 5) + (threadIdx.x >> 5);
  int lane = threadIdx.x & 31;
  int e0   = wave * EPW;
  if (e0 >= E) return;

  int rr[EPW], cc[EPW];
  int cnt = 0;
#pragma unroll
  for (int j = 0; j < EPW; ++j) {
    int e = e0 + j;
    if (e < E) {
      rr[j] = ei[e];
      cc[j] = ei[E + e];
      // spread lanes across the source row's cachelines (global_prefetch_b8)
      __builtin_prefetch(h + (size_t)cc[j] * dim + ((lane * dim) >> 5), 0, 1);
      cnt = j + 1;
    }
  }
#pragma unroll
  for (int j = 0; j < EPW; ++j) {
    if (j >= cnt) break;
    float norm = dinv[rr[j]] * dinv[cc[j]];
    const float* hs = h   + (size_t)cc[j] * dim;
    float*       ad = agg + (size_t)rr[j] * dim;
    for (int f = lane; f < dim; f += 32)
      atomicAdd(&ad[f], hs[f] * norm);
  }
}

// out_f32 = relu(agg + b); also emit bf16 copy for the next layer's GEMM A.
__global__ void k_bias_relu_cvt(const float* __restrict__ agg,
                                const float* __restrict__ b,
                                float* __restrict__ out_f32,
                                unsigned short* __restrict__ out_bf16,
                                int n, int dim) {
  int idx = blockIdx.x * blockDim.x + threadIdx.x;
  if (idx >= n * dim) return;
  int f = idx % dim;
  float v = agg[idx] + b[f];
  v = fmaxf(v, 0.0f);
  out_f32[idx]  = v;
  out_bf16[idx] = f32_to_bf16_bits(v);
}

// ------------------------------ pooling + head -----------------------------

__global__ void k_pool(const float* __restrict__ h4, const int* __restrict__ batch,
                       float* __restrict__ psum, float* __restrict__ pcnt, int n) {
  int idx = blockIdx.x * blockDim.x + threadIdx.x;      // over n*16
  if (idx >= n * 16) return;
  int i = idx >> 4;
  int f = idx & 15;
  int g = batch[i];
  atomicAdd(&psum[g * 16 + f], h4[idx]);
  if (f == 0) atomicAdd(&pcnt[g], 1.0f);
}

__global__ void k_head(const float* __restrict__ psum, const float* __restrict__ pcnt,
                       const float* __restrict__ Wp, const float* __restrict__ bp,
                       float* __restrict__ out, int n_graphs) {
  int idx = blockIdx.x * blockDim.x + threadIdx.x;      // over n_graphs*10
  if (idx >= n_graphs * 10) return;
  int g = idx / 10;
  int o = idx - g * 10;
  float inv = 1.0f / fmaxf(pcnt[g], 1.0f);
  float acc = bp[o];
#pragma unroll
  for (int f = 0; f < 16; ++f)
    acc += (psum[g * 16 + f] * inv) * Wp[f * 10 + o];
  out[idx] = acc;
}

// ------------------------------ launch -------------------------------------

static inline size_t align_up(size_t x, size_t a) { return (x + a - 1) & ~(a - 1); }

extern "C" void kernel_launch(void* const* d_in, const int* in_sizes, int n_in,
                              void* d_out, int out_size, void* d_ws, size_t ws_size,
                              hipStream_t stream) {
  (void)n_in; (void)out_size; (void)ws_size;
  const float* x     = (const float*)d_in[0];
  const int*   ei    = (const int*)d_in[1];
  const int*   batch = (const int*)d_in[2];
  const float* Wl[4] = {(const float*)d_in[3], (const float*)d_in[5],
                        (const float*)d_in[7], (const float*)d_in[9]};
  const float* bl[4] = {(const float*)d_in[4], (const float*)d_in[6],
                        (const float*)d_in[8], (const float*)d_in[10]};
  const float* Wp = (const float*)d_in[11];
  const float* bp = (const float*)d_in[12];
  float* out = (float*)d_out;

  const int N  = in_sizes[0] / 256;       // 100000 nodes
  const int E  = in_sizes[1] / 2;         // 3.2M edges
  const int NG = 512;
  const int dims[5] = {256, 128, 64, 32, 16};

  // workspace carve-out
  char* ws = (char*)d_ws;
  size_t cur = 0;
  unsigned short* feat_bf16 = (unsigned short*)(ws + cur);          // [N, <=256] bf16
  cur = align_up(cur + (size_t)N * 256 * sizeof(unsigned short), 256);
  float* h_pre = (float*)(ws + cur);                                // [N, <=128] f32
  cur = align_up(cur + (size_t)N * 128 * sizeof(float), 256);
  float* h_agg = (float*)(ws + cur);                                // [N, <=128] f32
  cur = align_up(cur + (size_t)N * 128 * sizeof(float), 256);
  unsigned short* Wt = (unsigned short*)(ws + cur);                 // [<=128, <=256] bf16
  cur = align_up(cur + (size_t)256 * 128 * sizeof(unsigned short), 256);
  float* deg  = (float*)(ws + cur); cur = align_up(cur + (size_t)N * sizeof(float), 256);
  float* dinv = (float*)(ws + cur); cur = align_up(cur + (size_t)N * sizeof(float), 256);
  float* psum = (float*)(ws + cur); cur = align_up(cur + (size_t)NG * 16 * sizeof(float), 256);
  float* pcnt = (float*)(ws + cur); cur = align_up(cur + (size_t)NG * sizeof(float), 256);

  const int TB = 256;
  auto blocks = [](long long n, int tb) { return (unsigned)((n + tb - 1) / tb); };

  // --- degrees / normalization -------------------------------------------
  k_fill<<<blocks(N, TB), TB, 0, stream>>>(deg, 1.0f, N);           // self loops
  k_degree<<<blocks(E, TB), TB, 0, stream>>>(ei, E, deg);
  k_dinv<<<blocks(N, TB), TB, 0, stream>>>(deg, dinv, N);

  // --- input features -> bf16 --------------------------------------------
  k_cvt_bf16<<<blocks((long long)N * 256, TB), TB, 0, stream>>>(x, feat_bf16, N * 256);

  // --- 4 GCN layers -------------------------------------------------------
  const int Mtiles = N / 16;                                        // 6250 exact
  const unsigned gemm_grid = (unsigned)((Mtiles + 3) / 4);
  for (int l = 0; l < 4; ++l) {
    const int K = dims[l], Nd = dims[l + 1];
    k_transpose_cvt<<<blocks((long long)K * Nd, TB), TB, 0, stream>>>(Wl[l], Wt, K, Nd);

    switch (l) {                                                    // (NT,K) dispatch
      case 0: k_wmma_gemm_fused<8, 256><<<gemm_grid, 128, 0, stream>>>(
                  feat_bf16, Wt, dinv, h_pre, h_agg, Mtiles); break;
      case 1: k_wmma_gemm_fused<4, 128><<<gemm_grid, 128, 0, stream>>>(
                  feat_bf16, Wt, dinv, h_pre, h_agg, Mtiles); break;
      case 2: k_wmma_gemm_fused<2, 64><<<gemm_grid, 128, 0, stream>>>(
                  feat_bf16, Wt, dinv, h_pre, h_agg, Mtiles); break;
      default: k_wmma_gemm_fused<1, 32><<<gemm_grid, 128, 0, stream>>>(
                  feat_bf16, Wt, dinv, h_pre, h_agg, Mtiles); break;
    }

    k_edge_scatter<<<blocks((E + 3) / 4, 8), TB, 0, stream>>>(ei, E, h_pre, dinv, h_agg, Nd);
    k_bias_relu_cvt<<<blocks((long long)N * Nd, TB), TB, 0, stream>>>(
        h_agg, bl[l], h_pre, feat_bf16, N, Nd);                     // f32 + bf16 outputs
  }

  // --- global mean pool + head -------------------------------------------
  k_fill<<<blocks(NG * 16, TB), TB, 0, stream>>>(psum, 0.0f, NG * 16);
  k_fill<<<blocks(NG, TB), TB, 0, stream>>>(pcnt, 0.0f, NG);
  k_pool<<<blocks((long long)N * 16, TB), TB, 0, stream>>>(h_pre, batch, psum, pcnt, N);
  k_head<<<blocks(NG * 10, TB), TB, 0, stream>>>(psum, pcnt, Wp, bp, out, NG);
}
